// Encoder_14293651161094
// MI455X (gfx1250) — compile-verified, matrix operand order checked by reading, not run on hardware
//
#include <hip/hip_runtime.h>
#include <hip/hip_bf16.h>

typedef _Float16 half_t;
typedef __attribute__((ext_vector_type(16))) _Float16 v16h;
typedef __attribute__((ext_vector_type(8)))  float    v8f;

#define T_LEN   80
#define N_AG    12
#define N_TOK   15360      // (16*12)*80
#define D_MODEL 128
#define N_HEAD  8
#define HC      1024
#define N_FRAME 1280       // 16*80
#define FB      2          // frames per workgroup in GAT (B-fragment reuse x2)
#define EPSV    1e-5f

// ---------------------------------------------------------------------------
// WMMA helpers (CDNA5 wave32, 16x16x32 f16 -> f32)
// ---------------------------------------------------------------------------
__device__ __forceinline__ v8f wmma_f16(v16h a, v16h b, v8f c) {
  return __builtin_amdgcn_wmma_f32_16x16x32_f16(false, a, false, b, (short)0, c,
                                                false, false);
}

// A fragment: 16x32 f16 tile from row-major LDS [16][lda].
// lane l: row m = l&15, k-base = (l>>4)*8; element e -> k = (e>>3)*16 + kb + (e&7)
__device__ __forceinline__ v16h load_a_frag(const half_t* A, int lda, int kt, int lane) {
  int m  = lane & 15;
  int kb = (lane >> 4) * 8;
  const half_t* p = A + m * lda + kt * 32;
  v16h a;
#pragma unroll
  for (int i = 0; i < 8; ++i) { a[i] = p[kb + i]; a[8 + i] = p[16 + kb + i]; }
  return a;
}

// B fragment: pre-swizzled in global memory, 32 lanes x 16 halves contiguous
__device__ __forceinline__ v16h load_b_frag(const half_t* Bs, int frag, int lane) {
  return *(const v16h*)(Bs + ((size_t)frag * 32 + lane) * 16);
}

__device__ __forceinline__ float wave_sum(float v) {
#pragma unroll
  for (int off = 16; off > 0; off >>= 1) v += __shfl_xor(v, off, 32);
  return v;
}

// ---------------------------------------------------------------------------
// Weight pre-swizzle: row-major f32 W[K x N] -> B-fragment f16 layout.
// frag = nt*(K/32) + kt ; lane n = l&15 ; element e -> k as in A layout.
// ---------------------------------------------------------------------------
__global__ __launch_bounds__(256) void swizzle_w(const float* __restrict__ W,
                                                 half_t* __restrict__ out,
                                                 int K, int N, int Kreal, int total) {
  for (int idx = blockIdx.x * 256 + threadIdx.x; idx < total;
       idx += gridDim.x * 256) {
    int e    = idx & 15;
    int l    = (idx >> 4) & 31;
    int frag = idx >> 9;
    int KT   = K >> 5;
    int kt   = frag % KT;
    int nt   = frag / KT;
    int n    = nt * 16 + (l & 15);
    int k    = kt * 32 + ((e >> 3) * 16) + ((l >> 4) * 8) + (e & 7);
    float v  = (k < Kreal) ? W[(size_t)k * N + n] : 0.f;
    out[idx] = (half_t)v;
  }
}

// ---------------------------------------------------------------------------
// K1: embeddings -> A f16 in LDS -> GEMM1 (K=64 padded from 36, N=64)
// 128 threads (4 waves), M-block 64, wave w = M-tile w, all 4 N-tiles.
// ---------------------------------------------------------------------------
__global__ __launch_bounds__(128) void k1_embed_gemm(
    const float* __restrict__ state_feat, const int* __restrict__ agent_ids,
    const float* __restrict__ hex_tab, const float* __restrict__ player_tab,
    const float* __restrict__ index_tab, const half_t* __restrict__ W1s,
    const float* __restrict__ b1, float* __restrict__ y1) {
  __shared__ half_t A[64][64];
  int m0  = blockIdx.x * 64;
  int tid = threadIdx.x;
  for (int idx = tid; idx < 64 * 64; idx += 128) {
    int r = idx >> 6, c = idx & 63;
    int p = m0 + r;
    float v = 0.f;
    if (c < 36) {
      int n = p / T_LEN;
      if (c < 12) {
        float hf = state_feat[p];
        int hid = (hf == 1e9f) ? 598 : (int)hf;
        if (hid < 0) hid = 0; if (hid > 599) hid = 599;
        v = hex_tab[hid * 12 + c];
      } else if (c < 24) {
        int aid = agent_ids[n]; if (aid < 0) aid = 0;
        v = player_tab[aid * 12 + (c - 12)];
      } else {
        v = index_tab[(n % N_AG) * 12 + (c - 24)];
      }
    }
    A[r][c] = (half_t)v;
  }
  __syncthreads();
  int wave = tid >> 5, lane = tid & 31;
  v16h af[2];
#pragma unroll
  for (int kt = 0; kt < 2; ++kt) af[kt] = load_a_frag(&A[wave * 16][0], 64, kt, lane);
#pragma unroll
  for (int nt = 0; nt < 4; ++nt) {
    v8f acc = {};
#pragma unroll
    for (int kt = 0; kt < 2; ++kt)
      acc = wmma_f16(af[kt], load_b_frag(W1s, nt * 2 + kt, lane), acc);
    int n = lane & 15, mh = (lane >> 4) * 8;
    float bias = b1[nt * 16 + n];
#pragma unroll
    for (int v = 0; v < 8; ++v) {
      int m = v + mh;
      y1[(size_t)(m0 + wave * 16 + m) * 64 + nt * 16 + n] = acc[v] + bias;
    }
  }
}

// ---------------------------------------------------------------------------
// Per-channel BatchNorm training stats (biased var): mean + rstd
// ---------------------------------------------------------------------------
__global__ __launch_bounds__(256) void stats_kernel(const float* __restrict__ y,
                                                    int C, float* __restrict__ mean,
                                                    float* __restrict__ rstd) {
  __shared__ float sh[256], sh2[256];
  int c = blockIdx.x;
  float s = 0.f, s2 = 0.f;
  for (int r = threadIdx.x; r < N_TOK; r += 256) {
    float v = y[(size_t)r * C + c];
    s += v; s2 += v * v;
  }
  sh[threadIdx.x] = s; sh2[threadIdx.x] = s2;
  __syncthreads();
  for (int off = 128; off > 0; off >>= 1) {
    if (threadIdx.x < off) {
      sh[threadIdx.x]  += sh[threadIdx.x + off];
      sh2[threadIdx.x] += sh2[threadIdx.x + off];
    }
    __syncthreads();
  }
  if (threadIdx.x == 0) {
    float m   = sh[0] * (1.f / N_TOK);
    float var = sh2[0] * (1.f / N_TOK) - m * m;
    mean[c] = m;
    rstd[c] = rsqrtf(var + EPSV);
  }
}

// ---------------------------------------------------------------------------
// K2: A = relu(bn1(y1)) [Mb=32,K=64] @ W2 [64,256] -> y2
// ---------------------------------------------------------------------------
__global__ __launch_bounds__(256) void k2_gemm(
    const float* __restrict__ y1, const float* __restrict__ mean1,
    const float* __restrict__ rstd1, const float* __restrict__ g1,
    const float* __restrict__ be1, const half_t* __restrict__ W2s,
    const float* __restrict__ b2, float* __restrict__ y2) {
  __shared__ half_t A[32][64];
  int m0 = blockIdx.x * 32, tid = threadIdx.x;
  for (int idx = tid; idx < 32 * 64; idx += 256) {
    int r = idx >> 6, c = idx & 63;
    float v = (y1[(size_t)(m0 + r) * 64 + c] - mean1[c]) * rstd1[c] * g1[c] + be1[c];
    A[r][c] = (half_t)(v > 0.f ? v : 0.f);
  }
  __syncthreads();
  int wave = tid >> 5, lane = tid & 31;
  int mt = wave & 1;
  v16h af[2];
#pragma unroll
  for (int kt = 0; kt < 2; ++kt) af[kt] = load_a_frag(&A[mt * 16][0], 64, kt, lane);
#pragma unroll
  for (int j = 0; j < 4; ++j) {
    int nt = (wave >> 1) * 4 + j;
    v8f acc = {};
#pragma unroll
    for (int kt = 0; kt < 2; ++kt)
      acc = wmma_f16(af[kt], load_b_frag(W2s, nt * 2 + kt, lane), acc);
    int n = lane & 15, mh = (lane >> 4) * 8;
    float bias = b2[nt * 16 + n];
#pragma unroll
    for (int v = 0; v < 8; ++v) {
      int m = v + mh;
      y2[(size_t)(m0 + mt * 16 + m) * 256 + nt * 16 + n] = acc[v] + bias;
    }
  }
}

// ---------------------------------------------------------------------------
// K3: A = relu(bn2(y2)) [Mb=32,K=256] @ W3 [256,128] -> y3
// ---------------------------------------------------------------------------
__global__ __launch_bounds__(256) void k3_gemm(
    const float* __restrict__ y2, const float* __restrict__ mean2,
    const float* __restrict__ rstd2, const float* __restrict__ g2,
    const float* __restrict__ be2, const half_t* __restrict__ W3s,
    const float* __restrict__ b3, float* __restrict__ y3) {
  __shared__ half_t A[32][256];
  int m0 = blockIdx.x * 32, tid = threadIdx.x;
  for (int idx = tid; idx < 32 * 256; idx += 256) {
    int r = idx >> 8, c = idx & 255;
    float v = (y2[(size_t)(m0 + r) * 256 + c] - mean2[c]) * rstd2[c] * g2[c] + be2[c];
    A[r][c] = (half_t)(v > 0.f ? v : 0.f);
  }
  __syncthreads();
  int wave = tid >> 5, lane = tid & 31;
  int mt = wave & 1;
  v16h af[8];
#pragma unroll
  for (int kt = 0; kt < 8; ++kt) af[kt] = load_a_frag(&A[mt * 16][0], 256, kt, lane);
#pragma unroll
  for (int j = 0; j < 2; ++j) {
    int nt = (wave >> 1) * 2 + j;
    v8f acc = {};
#pragma unroll
    for (int kt = 0; kt < 8; ++kt)
      acc = wmma_f16(af[kt], load_b_frag(W3s, nt * 8 + kt, lane), acc);
    int n = lane & 15, mh = (lane >> 4) * 8;
    float bias = b3[nt * 16 + n];
#pragma unroll
    for (int v = 0; v < 8; ++v) {
      int m = v + mh;
      y3[(size_t)(m0 + mt * 16 + m) * 128 + nt * 16 + n] = acc[v] + bias;
    }
  }
}

// x = relu(bn3(y3)) elementwise
__global__ __launch_bounds__(256) void bnrelu_kernel(
    const float* __restrict__ y, const float* __restrict__ mean,
    const float* __restrict__ rstd, const float* __restrict__ g,
    const float* __restrict__ be, float* __restrict__ x, int total) {
  int i = blockIdx.x * 256 + threadIdx.x;
  if (i < total) {
    int c = i & (D_MODEL - 1);
    float v = (y[i] - mean[c]) * rstd[c] * g[c] + be[c];
    x[i] = v > 0.f ? v : 0.f;
  }
}

// ---------------------------------------------------------------------------
// GATv2 layer: one workgroup per FB frames. 8 waves, wave h = head h.
//  - WMMA xl/xr projections; each B fragment loaded once, reused FB times
//    (halves L2 weight traffic, the dominant byte stream of this model)
//  - uses CDNA5 large-LDS: ~120 KB static shared per WG (320 KB WGP budget)
//  - attention + softmax per dst (11 edges + self-loop, edge mean fill)
//  - head mean + bout + residual + LayerNorm, fused.
// ---------------------------------------------------------------------------
__global__ __launch_bounds__(256) void gat_kernel(
    const float* __restrict__ xin, float* __restrict__ xout,
    const half_t* __restrict__ Wls, const half_t* __restrict__ Wrs,
    const float* __restrict__ bl, const float* __restrict__ br,
    const float* __restrict__ att, const float* __restrict__ Wev,
    const float* __restrict__ bout, const float* __restrict__ lng,
    const float* __restrict__ lnb, const float* __restrict__ edge_attr) {
  __shared__ half_t Alds[FB][16][128];        // 8 KB
  __shared__ half_t XL[FB][12][HC + 8];       // 49.5 KB
  __shared__ half_t XR[FB][12][HC + 8];       // 49.5 KB
  __shared__ float  outacc[FB][12][128];      // 12 KB
  __shared__ float  eaF[FB][144];             // 1.1 KB   (total ~120.7 KB)

  int f0  = blockIdx.x * FB;
  int tid = threadIdx.x, wave = tid >> 5, lane = tid & 31;

  // Phase 0: gather frame features -> f16 A ; edge_attr ; zero accumulators
#pragma unroll
  for (int ff = 0; ff < FB; ++ff) {
    int f = f0 + ff;
    int b = f / T_LEN, t = f - b * T_LEN;
    for (int idx = tid; idx < 16 * 128; idx += 256) {
      int r = idx >> 7, c = idx & 127;
      float v = 0.f;
      if (r < N_AG) v = xin[((size_t)(b * N_AG + r) * T_LEN + t) * D_MODEL + c];
      Alds[ff][r][c] = (half_t)v;
    }
    if (tid < 132) eaF[ff][tid] = edge_attr[(size_t)f * 132 + tid];
    for (int idx = tid; idx < 12 * 128; idx += 256) (&outacc[ff][0][0])[idx] = 0.f;
  }
  __syncthreads();
  if (tid < FB) {  // self-loop edge attr = mean over 132 real edges
    float s = 0.f;
    for (int e = 0; e < 132; ++e) s += eaF[tid][e];
    s *= (1.f / 132.f);
    for (int e = 0; e < 12; ++e) eaF[tid][132 + e] = s;
  }

  // Phase 1: xl = A @ Wl, xr = A @ Wr for this head's 128 columns.
  // B fragments hoisted over the frame loop: 1 load feeds FB WMMAs.
  int h = wave;
  v16h af[FB][4];
#pragma unroll
  for (int ff = 0; ff < FB; ++ff)
#pragma unroll
    for (int kt = 0; kt < 4; ++kt)
      af[ff][kt] = load_a_frag(&Alds[ff][0][0], 128, kt, lane);
#pragma unroll
  for (int nt = 0; nt < 8; ++nt) {
    int ntg = h * 8 + nt;
    v8f accL[FB], accR[FB];
#pragma unroll
    for (int ff = 0; ff < FB; ++ff) { accL[ff] = (v8f){}; accR[ff] = (v8f){}; }
#pragma unroll
    for (int kt = 0; kt < 4; ++kt) {
      v16h bL = load_b_frag(Wls, ntg * 4 + kt, lane);
      v16h bR = load_b_frag(Wrs, ntg * 4 + kt, lane);
#pragma unroll
      for (int ff = 0; ff < FB; ++ff) {
        accL[ff] = wmma_f16(af[ff][kt], bL, accL[ff]);
        accR[ff] = wmma_f16(af[ff][kt], bR, accR[ff]);
      }
    }
    int n = lane & 15, mh = (lane >> 4) * 8;
    float biasL = bl[ntg * 16 + n], biasR = br[ntg * 16 + n];
#pragma unroll
    for (int ff = 0; ff < FB; ++ff)
#pragma unroll
      for (int v = 0; v < 8; ++v) {
        int m = v + mh;
        if (m < N_AG) {
          XL[ff][m][ntg * 16 + n] = (half_t)(accL[ff][v] + biasL);
          XR[ff][m][ntg * 16 + n] = (half_t)(accR[ff][v] + biasR);
        }
      }
  }
  __syncthreads();

  // Phase 2: attention for head h (12 dsts x 12 incoming edges each)
  const float* atth = att + h * 128;
  const float* weh  = Wev + h * 128;
  for (int ff = 0; ff < FB; ++ff) {
    for (int d = 0; d < N_AG; ++d) {
      float sv[12];
      for (int e = 0; e < 12; ++e) {
        int s, eidx;
        if (e < 11) { s = e + (e >= d); eidx = s * 11 + d - (d > s); }
        else        { s = d; eidx = 132 + d; }
        float ea = eaF[ff][eidx];
        float part = 0.f;
#pragma unroll
        for (int cc = 0; cc < 4; ++cc) {
          int c = lane + cc * 32;
          float mv = (float)XL[ff][s][h * 128 + c] + (float)XR[ff][d][h * 128 + c] +
                     ea * weh[c];
          mv = mv > 0.f ? mv : 0.2f * mv;        // leaky_relu(0.2)
          part += mv * atth[c];
        }
        sv[e] = wave_sum(part);
      }
      float mx = sv[0];
#pragma unroll
      for (int e = 1; e < 12; ++e) mx = fmaxf(mx, sv[e]);
      float ssum = 0.f;
#pragma unroll
      for (int e = 0; e < 12; ++e) { sv[e] = __expf(sv[e] - mx); ssum += sv[e]; }
      float inv = 1.f / ssum;
      float oc[4] = {0.f, 0.f, 0.f, 0.f};
      for (int e = 0; e < 12; ++e) {
        int s = (e < 11) ? e + (e >= d) : d;
        float al = sv[e] * inv;
#pragma unroll
        for (int cc = 0; cc < 4; ++cc)
          oc[cc] += al * (float)XL[ff][s][h * 128 + lane + cc * 32];
      }
#pragma unroll
      for (int cc = 0; cc < 4; ++cc)
        atomicAdd(&outacc[ff][d][lane + cc * 32], oc[cc] * (1.f / N_HEAD));
    }
  }
  __syncthreads();

  // Phase 3: residual + LayerNorm, write out (in-place safe: row-private)
  for (int ff = 0; ff < FB; ++ff) {
    int f = f0 + ff;
    int b = f / T_LEN, t = f - b * T_LEN;
    for (int d = wave; d < N_AG; d += 8) {
      size_t base = ((size_t)(b * N_AG + d) * T_LEN + t) * D_MODEL;
      float yv[4];
      float s = 0.f;
#pragma unroll
      for (int cc = 0; cc < 4; ++cc) {
        int c = lane + cc * 32;
        float v = outacc[ff][d][c] + bout[c] + xin[base + c];
        yv[cc] = v; s += v;
      }
      float mean = wave_sum(s) * (1.f / D_MODEL);
      float s2 = 0.f;
#pragma unroll
      for (int cc = 0; cc < 4; ++cc) { float dv = yv[cc] - mean; s2 += dv * dv; }
      float rstd = rsqrtf(wave_sum(s2) * (1.f / D_MODEL) + EPSV);
#pragma unroll
      for (int cc = 0; cc < 4; ++cc) {
        int c = lane + cc * 32;
        xout[base + c] = (yv[cc] - mean) * rstd * lng[c] + lnb[c];
      }
    }
  }
}

// ---------------------------------------------------------------------------
// Host launcher
// ---------------------------------------------------------------------------
extern "C" void kernel_launch(void* const* d_in, const int* in_sizes, int n_in,
                              void* d_out, int out_size, void* d_ws, size_t ws_size,
                              hipStream_t stream) {
  const float* state_feat = (const float*)d_in[0];
  const int*   agent_ids  = (const int*)d_in[2];
  const float* edge_attr  = (const float*)d_in[4];
  const float* hex_tab    = (const float*)d_in[5];
  const float* player_tab = (const float*)d_in[6];
  const float* index_tab  = (const float*)d_in[7];
  const float* W1 = (const float*)d_in[8];
  const float* b1 = (const float*)d_in[9];
  const float* g1 = (const float*)d_in[10];
  const float* be1 = (const float*)d_in[11];
  const float* W2 = (const float*)d_in[12];
  const float* b2 = (const float*)d_in[13];
  const float* g2 = (const float*)d_in[14];
  const float* be2 = (const float*)d_in[15];
  const float* W3 = (const float*)d_in[16];
  const float* b3 = (const float*)d_in[17];
  const float* g3 = (const float*)d_in[18];
  const float* be3 = (const float*)d_in[19];
  const float* Wl = (const float*)d_in[20];
  const float* bl = (const float*)d_in[21];
  const float* Wr = (const float*)d_in[22];
  const float* br = (const float*)d_in[23];
  const float* att = (const float*)d_in[24];
  const float* Wev = (const float*)d_in[25];
  const float* bout = (const float*)d_in[26];
  const float* lng = (const float*)d_in[27];
  const float* lnb = (const float*)d_in[28];
  (void)in_sizes; (void)n_in; (void)out_size; (void)ws_size;

  // workspace carve (256B aligned)
  uint8_t* wp = (uint8_t*)d_ws;
  auto carve = [&](size_t bytes) {
    void* p = (void*)wp;
    wp += (bytes + 255) & ~(size_t)255;
    return p;
  };
  half_t* W1s = (half_t*)carve(2 * 4 * 512 * sizeof(half_t));     // K=64,N=64
  half_t* W2s = (half_t*)carve(2 * 16 * 512 * sizeof(half_t));    // K=64,N=256
  half_t* W3s = (half_t*)carve(8 * 8 * 512 * sizeof(half_t));     // K=256,N=128
  half_t* Wls[3]; half_t* Wrs[3];
  for (int i = 0; i < 3; ++i) Wls[i] = (half_t*)carve(4 * 64 * 512 * sizeof(half_t));
  for (int i = 0; i < 3; ++i) Wrs[i] = (half_t*)carve(4 * 64 * 512 * sizeof(half_t));
  float* y1 = (float*)carve((size_t)N_TOK * 64 * sizeof(float));
  float* y2 = (float*)carve((size_t)N_TOK * 256 * sizeof(float));
  float* y3 = (float*)carve((size_t)N_TOK * 128 * sizeof(float));
  float* x  = (float*)carve((size_t)N_TOK * 128 * sizeof(float));
  float* mean1 = (float*)carve(64 * sizeof(float));
  float* rstd1 = (float*)carve(64 * sizeof(float));
  float* mean2 = (float*)carve(256 * sizeof(float));
  float* rstd2 = (float*)carve(256 * sizeof(float));
  float* mean3 = (float*)carve(128 * sizeof(float));
  float* rstd3 = (float*)carve(128 * sizeof(float));

  // 1) pre-swizzle weights to WMMA B-fragment f16 layout
  auto swz = [&](const float* W, half_t* out, int K, int N, int Kreal) {
    int total = (K / 32) * (N / 16) * 512;
    swizzle_w<<<(total + 255) / 256, 256, 0, stream>>>(W, out, K, N, Kreal, total);
  };
  swz(W1, W1s, 64, 64, 36);
  swz(W2, W2s, 64, 256, 64);
  swz(W3, W3s, 256, 128, 256);
  for (int i = 0; i < 3; ++i) swz(Wl + (size_t)i * 128 * 1024, Wls[i], 128, 1024, 128);
  for (int i = 0; i < 3; ++i) swz(Wr + (size_t)i * 128 * 1024, Wrs[i], 128, 1024, 128);

  // 2) MLP with training-mode BatchNorm
  k1_embed_gemm<<<N_TOK / 64, 128, 0, stream>>>(state_feat, agent_ids, hex_tab,
                                                player_tab, index_tab, W1s, b1, y1);
  stats_kernel<<<64, 256, 0, stream>>>(y1, 64, mean1, rstd1);
  k2_gemm<<<N_TOK / 32, 256, 0, stream>>>(y1, mean1, rstd1, g1, be1, W2s, b2, y2);
  stats_kernel<<<256, 256, 0, stream>>>(y2, 256, mean2, rstd2);
  k3_gemm<<<N_TOK / 32, 256, 0, stream>>>(y2, mean2, rstd2, g2, be2, W3s, b3, y3);
  stats_kernel<<<128, 256, 0, stream>>>(y3, 128, mean3, rstd3);
  bnrelu_kernel<<<(N_TOK * 128 + 255) / 256, 256, 0, stream>>>(y3, mean3, rstd3, g3,
                                                               be3, x, N_TOK * 128);

  // 3) three GATv2 layers, fused attention + residual + LayerNorm
  for (int i = 0; i < 3; ++i) {
    const float* xi = x;
    float* xo = (i == 2) ? (float*)d_out : x;
    gat_kernel<<<N_FRAME / FB, 256, 0, stream>>>(
        xi, xo, Wls[i], Wrs[i], bl + (size_t)i * 1024, br + (size_t)i * 1024,
        att + (size_t)i * 8 * 128, Wev + (size_t)i * 1024, bout + (size_t)i * 128,
        lng + (size_t)i * 128, lnb + (size_t)i * 128, edge_attr);
  }
}
